// RoPEBERTAttention_12275016532406
// MI455X (gfx1250) — matrix operand order repaired and checked by reading, not compiled, not run on hardware
//
#include <hip/hip_runtime.h>
#include <hip/hip_bf16.h>

// ---------------------------------------------------------------------------
// RoPE-BERT attention, MI455X (gfx1250), wave32, v_wmma_f32_16x16x32_bf16.
// Pipeline:
//   1) fp32 -> bf16 conversion of X and the four weight matrices
//   2) Q/K/V projections as bf16 WMMA GEMMs with fused bias (+RoPE +1/sqrt(D)
//      for Q/K, transposed [B,H,D,S] store for V)
//   3) flash-attention per wave (16 q rows), computing S^T = K*Q^T so the
//      score tile lands directly in the A-operand layout for P*V
//   4) output projection GEMM -> fp32 d_out
// ---------------------------------------------------------------------------

typedef __attribute__((ext_vector_type(8)))  __bf16 v8bf;
typedef __attribute__((ext_vector_type(16))) __bf16 v16bf;
typedef __attribute__((ext_vector_type(8)))  float  v8f;

#define B_SZ   2
#define S_LEN  2048
#define E_DIM  1024
#define H_NUM  16
#define D_HEAD 64
#define M_TOK  (B_SZ * S_LEN)   // 4096 tokens

static __device__ __forceinline__ v8bf ld8(const __bf16* p) {
  return *(const v8bf*)p;
}
static __device__ __forceinline__ v16bf cat8(v8bf lo, v8bf hi) {
  v16bf r;
#pragma unroll
  for (int i = 0; i < 8; ++i) { r[i] = lo[i]; r[i + 8] = hi[i]; }
  return r;
}

// WGP-scope prefetch (scope field 0 = WGP: fills every cache level).
// __builtin_prefetch only reaches SE/DEV scope, which skip the WGP cache.
// Lead distance lives in the constant offset field so the address VGPRs are
// CSE'd with the adjacent global_load_b128.
static __device__ __forceinline__ void prefetch_l0_512(const void* p) {
  asm volatile("global_prefetch_b8 %0, off offset:512" :: "v"(p));
}
static __device__ __forceinline__ void prefetch_l0_4k(const void* p) {
  asm volatile("global_prefetch_b8 %0, off offset:4096" :: "v"(p));
}

// ---------------------------------------------------------------------------
// fp32 -> bf16 (vectorized x4)
// ---------------------------------------------------------------------------
__global__ void cvt_bf16(const float* __restrict__ in, __bf16* __restrict__ out, int n) {
  int i = (blockIdx.x * blockDim.x + threadIdx.x) * 4;
  if (i < n) {
    float4 f = *(const float4*)(in + i);
    out[i + 0] = (__bf16)f.x;
    out[i + 1] = (__bf16)f.y;
    out[i + 2] = (__bf16)f.z;
    out[i + 3] = (__bf16)f.w;
  }
}

// ---------------------------------------------------------------------------
// GEMM:  C[m,n] = sum_k A[m,k] * W[n,k] + bias[n]    (M=4096, N=K=1024)
// Block = 128 threads = 4 waves; block tile 64(M) x 128(N).
// Each wave owns a 32x64 tile = 2x4 WMMA tiles -> 8 WMMAs per 12 B128 loads
// per k-step of 32. Named (non-array) row pointers keep address-space
// inference intact -> global_load_b128 (LOADcnt-only waits), uniform k0
// folds into instruction offsets. Unroll capped at 2; launch_bounds(128,1)
// widens the VGPR budget so nothing spills.
// mode 0: Q  -> RoPE + 1/sqrt(D) scale, store bf16 [B,H,S,D]
// mode 1: K  -> RoPE,                 store bf16 [B,H,S,D]
// mode 2: V  -> transposed store bf16 [B,H,D,S]
// mode 3: O  -> fp32 store [M,N] (final output)
// ---------------------------------------------------------------------------
__global__ void __launch_bounds__(128, 1)
gemm_qkvo(const __bf16* __restrict__ A, const __bf16* __restrict__ W,
          const float* __restrict__ bias, __bf16* __restrict__ outb,
          float* __restrict__ outf, int mode) {
  const int lane = threadIdx.x & 31;
  const int wave = threadIdx.x >> 5;
  const int lm   = lane & 15;     // lane within half
  const int lh   = lane >> 4;     // lane half (0/1)

  const int m_base = blockIdx.x * 64  + (wave >> 1) * 32;
  const int n_base = blockIdx.y * 128 + (wave & 1) * 64;

  // Named row base pointers (A layout: K 0..15 | 16..31 per lane half;
  // B layout: K 0..15 | 16..31 per lane half).
  const __bf16* pA0 = A + (size_t)(m_base + lm) * E_DIM + lh * 16;
  const __bf16* pA1 = pA0 + 16 * E_DIM;
  const __bf16* pB0 = W + (size_t)(n_base + lm) * E_DIM + lh * 16;
  const __bf16* pB1 = pB0 + 16 * E_DIM;
  const __bf16* pB2 = pB0 + 32 * E_DIM;
  const __bf16* pB3 = pB0 + 48 * E_DIM;

  v8f c[2][4] = {};

#pragma unroll 2
  for (int k0 = 0; k0 < E_DIM; k0 += 32) {
    const bool pf = (k0 & 63) == 0;  // prefetch ~512B ahead, every other step
    if (pf) {
      prefetch_l0_512(pA0 + k0);
      prefetch_l0_512(pA1 + k0);
      prefetch_l0_512(pB0 + k0);
      prefetch_l0_512(pB1 + k0);
      prefetch_l0_512(pB2 + k0);
      prefetch_l0_512(pB3 + k0);
    }
    v16bf av[2], bv[4];
    av[0] = cat8(ld8(pA0 + k0), ld8(pA0 + k0 + 8));
    av[1] = cat8(ld8(pA1 + k0), ld8(pA1 + k0 + 8));
    bv[0] = cat8(ld8(pB0 + k0), ld8(pB0 + k0 + 8));
    bv[1] = cat8(ld8(pB1 + k0), ld8(pB1 + k0 + 8));
    bv[2] = cat8(ld8(pB2 + k0), ld8(pB2 + k0 + 8));
    bv[3] = cat8(ld8(pB3 + k0), ld8(pB3 + k0 + 8));
#pragma unroll
    for (int i = 0; i < 2; ++i)
#pragma unroll
      for (int j = 0; j < 4; ++j)
        c[i][j] = __builtin_amdgcn_wmma_f32_16x16x32_bf16(
            false, av[i], false, bv[j], (short)0, c[i][j], false, false);
  }

  // Epilogue. C layout: lane = N-column; VGPR r -> row (r + lh*8) of the tile.
#pragma unroll
  for (int i = 0; i < 2; ++i) {
#pragma unroll
    for (int j = 0; j < 4; ++j) {
      const int n  = n_base + j * 16 + lm;  // output feature
      const int h  = n >> 6;                // head (D=64)
      const int d  = n & 63;                // dim within head
      const float bn = bias[n];
      const float invf = __expf(-0.14391156f * (float)(d & ~1)); // theta^(-2i/D)
#pragma unroll
      for (int r = 0; r < 8; ++r) {
        const int m = m_base + i * 16 + r + lh * 8;  // token row
        float v = c[i][j][r] + bn;
        if (mode <= 1) {
          // RoPE: pairs are adjacent features -> adjacent lanes.
          const int pos = m & (S_LEN - 1);
          float sn, cs;
          __sincosf((float)pos * invf, &sn, &cs);
          const float pv = __shfl_xor(v, 1, 32);
          float rv = (d & 1) ? (v * cs + pv * sn) : (v * cs - pv * sn);
          if (mode == 0) rv *= 0.125f;  // 1/sqrt(64) folded into Q
          const int b = m >> 11, s = m & (S_LEN - 1);
          outb[((size_t)(b * H_NUM + h) * S_LEN + s) * D_HEAD + d] = (__bf16)rv;
        } else if (mode == 2) {
          const int b = m >> 11, s = m & (S_LEN - 1);
          // transposed: [B,H,D,S] so attention's P*V B-operand is contiguous
          outb[((size_t)(b * H_NUM + h) * D_HEAD + d) * S_LEN + s] = (__bf16)v;
        } else {
          outf[(size_t)m * E_DIM + n] = v;
        }
      }
    }
  }
}

// ---------------------------------------------------------------------------
// Flash attention, one wave = 16 query rows x full head (D=64).
// Computes S^T = K * Q^T per 32-kv chunk: the C tile (lane=q, VGPR=kv) is
// exactly the A-operand layout required by P*V, so no transpose is needed.
// Online softmax in fp32; output written bf16 token-major [B*S, E].
// ---------------------------------------------------------------------------
__global__ void attn_fa(const __bf16* __restrict__ q, const __bf16* __restrict__ k,
                        const __bf16* __restrict__ vt, __bf16* __restrict__ aout) {
  const int lane = threadIdx.x & 31;
  const int lm   = lane & 15;
  const int lh   = lane >> 4;

  const int wid    = blockIdx.x * (blockDim.x >> 5) + (threadIdx.x >> 5);
  const int qtiles = S_LEN >> 4;           // 128
  const int qt = wid % qtiles;
  const int bh = wid / qtiles;             // 0..31
  const int q0 = qt * 16;

  const __bf16* qp = q  + (size_t)bh * S_LEN * D_HEAD;
  const __bf16* kp = k  + (size_t)bh * S_LEN * D_HEAD;
  const __bf16* vp = vt + (size_t)bh * D_HEAD * S_LEN;

  // Q as B-operand (contraction over d): lane = q column, K split 8/8 per half (A split).
  const __bf16* qr = qp + (size_t)(q0 + lm) * D_HEAD;
  const v16bf qb0 = cat8(ld8(qr + lh * 8),      ld8(qr + lh * 8 + 16));       // d 0..31
  const v16bf qb1 = cat8(ld8(qr + 32 + lh * 8), ld8(qr + 48 + lh * 8));  // d 32..63

  float mrun = -1e30f, lrun = 0.f;
  v8f o[4] = {};   // 16(q) x 64(d) fp32 accumulator, lane = d column

  for (int kv0 = 0; kv0 < S_LEN; kv0 += 32) {
    // scores^T: two 16(kv) x 16(q) tiles; K rows are the A operand.
    v8f st[2];
#pragma unroll
    for (int t = 0; t < 2; ++t) {
      const __bf16* kr = kp + (size_t)(kv0 + t * 16 + lm) * D_HEAD + lh * 8;
      prefetch_l0_4k(kr);                                // next kv chunk
      const v16bf a0 = cat8(ld8(kr),      ld8(kr + 16));   // d 0..31  (A split)
      const v16bf a1 = cat8(ld8(kr + 32), ld8(kr + 48));   // d 32..63
      v8f z = {};
      z = __builtin_amdgcn_wmma_f32_16x16x32_bf16(false, a0, false, qb0, (short)0, z, false, false);
      z = __builtin_amdgcn_wmma_f32_16x16x32_bf16(false, a1, false, qb1, (short)0, z, false, false);
      st[t] = z;
    }

    // Online softmax. Each lane holds 16 kv scores of its q; partner lane
    // (xor 16) holds the other 16 -> one shfl for the chunk max.
    float cmax = -1e30f;
#pragma unroll
    for (int t = 0; t < 2; ++t)
#pragma unroll
      for (int r = 0; r < 8; ++r) cmax = fmaxf(cmax, st[t][r]);
    cmax = fmaxf(cmax, __shfl_xor(cmax, 16, 32));
    const float mnew = fmaxf(mrun, cmax);
    const float corr = __expf(mrun - mnew);

    float p[16];
    float psum = 0.f;
#pragma unroll
    for (int t = 0; t < 2; ++t)
#pragma unroll
      for (int r = 0; r < 8; ++r) {
        const float e = __expf(st[t][r] - mnew);
        p[t * 8 + r] = e;
        psum += e;
      }
    lrun = lrun * corr + psum;
    mrun = mnew;

    // Rescale accumulator: o VGPR r belongs to q row (r + lh*8); the factor
    // for q row x lives in lane x (ds_bpermute broadcast).
#pragma unroll
    for (int r = 0; r < 8; ++r) {
      const float f = __shfl(corr, r + lh * 8, 32);
#pragma unroll
      for (int j = 0; j < 4; ++j) o[j][r] *= f;
    }

    // P as A operand: st layout already matches the 16x32 bf16 A layout.
    v16bf pa;
#pragma unroll
    for (int e = 0; e < 16; ++e) pa[e] = (__bf16)p[e];

    // P * V: B operand from transposed V (rows contiguous in kv).
#pragma unroll
    for (int j = 0; j < 4; ++j) {
      const __bf16* vr = vp + (size_t)(j * 16 + lm) * S_LEN + kv0 + lh * 8;
      const v16bf vb = cat8(ld8(vr), ld8(vr + 16));
      o[j] = __builtin_amdgcn_wmma_f32_16x16x32_bf16(false, pa, false, vb, (short)0, o[j], false, false);
    }
  }

  // Finalize: 1/l per q row, store bf16 token-major [B*S, E].
  const float ltot = lrun + __shfl_xor(lrun, 16, 32);
  const float linv = 1.f / ltot;
  const int b = bh / H_NUM, h = bh % H_NUM;
#pragma unroll
  for (int r = 0; r < 8; ++r) {
    const float f = __shfl(linv, r + lh * 8, 32);
    const int qrow = q0 + r + lh * 8;
    __bf16* dst = aout + (size_t)(b * S_LEN + qrow) * E_DIM + h * D_HEAD + lm;
#pragma unroll
    for (int j = 0; j < 4; ++j) dst[j * 16] = (__bf16)(o[j][r] * f);
  }
}

// ---------------------------------------------------------------------------
// Host launcher
// ---------------------------------------------------------------------------
extern "C" void kernel_launch(void* const* d_in, const int* in_sizes, int n_in,
                              void* d_out, int out_size, void* d_ws, size_t ws_size,
                              hipStream_t stream) {
  const float* X  = (const float*)d_in[0];
  const float* Wq = (const float*)d_in[1];
  const float* bq = (const float*)d_in[2];
  const float* Wk = (const float*)d_in[3];
  const float* bk = (const float*)d_in[4];
  const float* Wv = (const float*)d_in[5];
  const float* bv = (const float*)d_in[6];
  const float* Wo = (const float*)d_in[7];
  const float* bo = (const float*)d_in[8];
  float* out = (float*)d_out;

  char* ws = (char*)d_ws;
  const size_t MB = 1024 * 1024;
  __bf16* Xb  = (__bf16*)(ws);             //  8 MB  [4096,1024]
  __bf16* Wqb = (__bf16*)(ws + 8  * MB);   //  2 MB
  __bf16* Wkb = (__bf16*)(ws + 10 * MB);   //  2 MB
  __bf16* Wvb = (__bf16*)(ws + 12 * MB);   //  2 MB
  __bf16* Wob = (__bf16*)(ws + 14 * MB);   //  2 MB
  __bf16* Qh  = (__bf16*)(ws + 16 * MB);   //  8 MB  [B,H,S,D]
  __bf16* Kh  = (__bf16*)(ws + 24 * MB);   //  8 MB  [B,H,S,D]
  __bf16* Vt  = (__bf16*)(ws + 32 * MB);   //  8 MB  [B,H,D,S]
  __bf16* Ab  = (__bf16*)(ws);             //  8 MB  [B*S,E], aliases Xb
                                           //  (X consumed before attn runs)

  // 1) fp32 -> bf16
  cvt_bf16<<<(M_TOK * E_DIM / 4 + 255) / 256, 256, 0, stream>>>(X, Xb, M_TOK * E_DIM);
  cvt_bf16<<<(E_DIM * E_DIM / 4 + 255) / 256, 256, 0, stream>>>(Wq, Wqb, E_DIM * E_DIM);
  cvt_bf16<<<(E_DIM * E_DIM / 4 + 255) / 256, 256, 0, stream>>>(Wk, Wkb, E_DIM * E_DIM);
  cvt_bf16<<<(E_DIM * E_DIM / 4 + 255) / 256, 256, 0, stream>>>(Wv, Wvb, E_DIM * E_DIM);
  cvt_bf16<<<(E_DIM * E_DIM / 4 + 255) / 256, 256, 0, stream>>>(Wo, Wob, E_DIM * E_DIM);

  // 2) projections (RoPE+scale fused for Q, RoPE for K, transposed store for V)
  dim3 gg(M_TOK / 64, E_DIM / 128);
  gemm_qkvo<<<gg, 128, 0, stream>>>(Xb, Wqb, bq, Qh, nullptr, 0);
  gemm_qkvo<<<gg, 128, 0, stream>>>(Xb, Wkb, bk, Kh, nullptr, 1);
  gemm_qkvo<<<gg, 128, 0, stream>>>(Xb, Wvb, bv, Vt, nullptr, 2);

  // 3) flash attention: one wave per 16 query rows; 4096 waves total
  attn_fa<<<(B_SZ * H_NUM * (S_LEN / 16)) / 4, 128, 0, stream>>>(Qh, Kh, Vt, Ab);

  // 4) output projection -> fp32
  gemm_qkvo<<<gg, 128, 0, stream>>>(Ab, Wob, bo, nullptr, out, 3);
}
